// MultiHeadAttention_75505525064373
// MI455X (gfx1250) — compile-verified, matrix operand order checked by reading, not run on hardware
//
#include <hip/hip_runtime.h>

// ---------------------------------------------------------------------------
// Spiking MHA forward for MI455X (gfx1250, wave32, WMMA f32_16x16x32_f16).
// T=4, B=8, L=512, D=512, H=8, dk=dv=64.
// ---------------------------------------------------------------------------

typedef __attribute__((ext_vector_type(16))) _Float16 v16h;
typedef __attribute__((ext_vector_type(8)))  float    v8f;
typedef __attribute__((ext_vector_type(4)))  unsigned int v4u;

#define DEV __device__ __forceinline__

constexpr int   T_ = 4, B_ = 8, L_ = 512, D_ = 512, H_ = 8, DKV = 64;
constexpr int   M_ = T_ * B_ * L_;   // 16384 rows for the big GEMMs
constexpr int   C_ = H_ * DKV;       // 512 channels
constexpr float INV_TEMP = 0.125f;   // 1/sqrt(64)
constexpr float EPS_ = 1e-5f;

// ---- WMMA fragment helpers -------------------------------------------------
// f16 16x32 A/B fragment from a row-major [16 rows x K] slab, row stride `ld`
// (in halves). ISA layout: lanes 0-15 hold K 0..7 & 16..23, lanes 16-31 hold
// K 8..15 & 24..31; row = lane&15. Two 16-byte loads per lane.
union FragU { v16h h; struct { v4u lo, hi; } u; };

DEV v16h load_frag16(const _Float16* __restrict__ src, int ld) {
  const int lane = threadIdx.x & 31;
  const int row  = lane & 15;
  const int koff = (lane >> 4) << 3;       // 0 or 8
  const _Float16* p = src + row * ld + koff;
  FragU f;
  f.u.lo = *(const v4u*)(p);
  f.u.hi = *(const v4u*)(p + 16);
  return f.h;
}

DEV v8f wmma_f16(v16h a, v16h b, v8f c) {
  return __builtin_amdgcn_wmma_f32_16x16x32_f16(
      /*neg_a=*/false, a, /*neg_b=*/false, b,
      /*c_mod=*/(short)0, c, /*reuse_a=*/false, /*reuse_b=*/false);
}

// ---- 0) fp32 -> fp16 conversion -------------------------------------------
__global__ __launch_bounds__(256)
void cvt_f32_f16(const float* __restrict__ s, _Float16* __restrict__ d, int n) {
  int i = blockIdx.x * 256 + threadIdx.x;
  if (i < n) d[i] = (_Float16)s[i];
}

// ---- 1) Projection GEMM: C[M,512] = A[M,512] * W[512,512]^T + bias ---------
// Block = 16 rows x 512 cols, 8 waves; wave w handles 4 N-tiles (64 cols).
__global__ __launch_bounds__(256)
void proj_gemm(const _Float16* __restrict__ A, const _Float16* __restrict__ W,
               const float* __restrict__ bias, float* __restrict__ Cout) {
  const int m0   = blockIdx.x * 16;
  const int wave = threadIdx.x >> 5;
  const int lane = threadIdx.x & 31;
  const int n0   = wave * 64;
  v8f acc[4] = {};
  for (int k = 0; k < 512; k += 32) {
    v16h a = load_frag16(A + (long)m0 * 512 + k, 512);
#pragma unroll
    for (int t = 0; t < 4; ++t) {
      v16h b = load_frag16(W + (long)(n0 + t * 16) * 512 + k, 512);
      acc[t] = wmma_f16(a, b, acc[t]);
    }
  }
  const int rsel = (lane >> 4) << 3;
  const int col  = lane & 15;
#pragma unroll
  for (int t = 0; t < 4; ++t) {
    const int n = n0 + t * 16 + col;
    const float bn = bias[n];
#pragma unroll
    for (int r = 0; r < 8; ++r)
      Cout[(long)(m0 + r + rsel) * 512 + n] = acc[t][r] + bn;
  }
}

// ---- 2) BatchNorm stats per channel (train-mode biased variance) ----------
__global__ __launch_bounds__(256)
void bn_stats(const float* __restrict__ X, const float* __restrict__ g,
              const float* __restrict__ b, float* __restrict__ scaleOut,
              float* __restrict__ shiftOut) {
  const int c = blockIdx.x;
  float s = 0.f, ss = 0.f;
  for (int m = threadIdx.x; m < M_; m += 256) {
    float v = X[(long)m * 512 + c];
    s += v; ss += v * v;
  }
  __shared__ float rs[8], rss[8];
  for (int o = 16; o > 0; o >>= 1) {
    s  += __shfl_down(s, o, 32);
    ss += __shfl_down(ss, o, 32);
  }
  if ((threadIdx.x & 31) == 0) { rs[threadIdx.x >> 5] = s; rss[threadIdx.x >> 5] = ss; }
  __syncthreads();
  if (threadIdx.x == 0) {
    float S = 0.f, SS = 0.f;
    for (int i = 0; i < 8; ++i) { S += rs[i]; SS += rss[i]; }
    const float mean = S / (float)M_;
    const float var  = SS / (float)M_ - mean * mean;
    const float sc   = g[c] * rsqrtf(var + EPS_);
    scaleOut[c] = sc;
    shiftOut[c] = b[c] - mean * sc;
  }
}

// ---- 3) Fused BN-apply + LIF over T + spike emit ---------------------------
// One thread per (b,l,c); scans T sequentially. Spikes written as f16 0/1.
// transposed==0 -> [T,H,B,L,64] (Q/K layout); ==1 -> [T,H,B,64,L] (V^T).
__global__ __launch_bounds__(256)
void lif_spikes(const float* __restrict__ X, const float* __restrict__ scale,
                const float* __restrict__ shift, _Float16* __restrict__ out,
                int transposed) {
  const int idx = blockIdx.x * 256 + threadIdx.x;        // over B*L*512
  if (idx >= B_ * L_ * C_) return;
  const int c = idx & 511;
  const int l = (idx >> 9) & 511;
  const int b = idx >> 18;
  const int h = c >> 6, d = c & 63;
  const float sc = scale[c], sh = shift[c];
  float vm = 0.f;
#pragma unroll
  for (int t = 0; t < T_; ++t) {
    const float x = X[((long)(t * B_ + b) * L_ + l) * C_ + c] * sc + sh;
    vm += (x - vm) * 0.5f;                  // tau = 2
    const float sp = (vm >= 1.f) ? 1.f : 0.f;
    vm *= (1.f - sp);                       // hard reset
    const long hb = (long)(t * H_ + h) * B_ + b;
    if (!transposed) out[(hb * L_ + l) * 64 + d] = (_Float16)sp;
    else             out[(hb * 64 + d) * L_ + l] = (_Float16)sp;
  }
}

// ---- 4) Fused attention: QK^T (WMMA) -> mask -> softmax -> PV (WMMA) -------
// Block = one 16-query strip of one (t,h,b). 4 waves / 128 threads.
// Score strip (16x512 f32) and prob strip (16x512 f16) live in LDS.
__global__ __launch_bounds__(128)
void attention(const _Float16* __restrict__ QS,   // [T,H,B,L,64]
               const _Float16* __restrict__ KS,   // [T,H,B,L,64]
               const _Float16* __restrict__ VT,   // [T,H,B,64,L]
               const int* __restrict__ mask,      // [B,1,L]
               float* __restrict__ attnOut,       // [T,H,B,L,L] (into d_out)
               _Float16* __restrict__ ctxOut) {   // [T,B,L,512] merged heads
  __shared__ __align__(16) float    sS[16][512];  // 32 KB
  __shared__ __align__(16) _Float16 sP[16][512];  // 16 KB
  const int t    = blockIdx.z;
  const int h    = blockIdx.y >> 3, b = blockIdx.y & 7;
  const int q0   = blockIdx.x * 16;
  const int wave = threadIdx.x >> 5, lane = threadIdx.x & 31;
  const long hb  = (long)(t * H_ + h) * B_ + b;
  const int rsel = (lane >> 4) << 3;
  const int col  = lane & 15;

  // --- scores = (qs @ ks^T) / sqrt(dk), masked ---
  const _Float16* qbase = QS + (hb * L_ + q0) * 64;
  const v16h a0 = load_frag16(qbase, 64);
  const v16h a1 = load_frag16(qbase + 32, 64);
  for (int nt = wave; nt < 32; nt += 4) {
    const int k0 = nt * 16;
    const _Float16* kbase = KS + (hb * L_ + k0) * 64;
    v8f c = {};
    c = wmma_f16(a0, load_frag16(kbase, 64), c);
    c = wmma_f16(a1, load_frag16(kbase + 32, 64), c);
    const int  kc  = k0 + col;
    const bool msk = mask[b * L_ + kc] != 0;
#pragma unroll
    for (int r = 0; r < 8; ++r)
      sS[r + rsel][kc] = msk ? -1e9f : c[r] * INV_TEMP;
  }
  __syncthreads();

  // --- row softmax: 16 groups of 8 threads, width-8 shuffle reductions ---
  {
    const int row = threadIdx.x >> 3;
    const int sub = threadIdx.x & 7;
    float* rowp = &sS[row][sub * 64];
    float mx = -1e30f;
    for (int e = 0; e < 64; ++e) mx = fmaxf(mx, rowp[e]);
    for (int o = 4; o > 0; o >>= 1) mx = fmaxf(mx, __shfl_xor(mx, o, 8));
    float sum = 0.f;
    for (int e = 0; e < 64; ++e) { float ev = __expf(rowp[e] - mx); rowp[e] = ev; sum += ev; }
    for (int o = 4; o > 0; o >>= 1) sum += __shfl_xor(sum, o, 8);
    const float inv = 1.f / sum;
    float*     gout = attnOut + (hb * L_ + (q0 + row)) * L_ + sub * 64;
    _Float16*  pp   = &sP[row][sub * 64];
    for (int e = 0; e < 64; ++e) {
      const float p = rowp[e] * inv;
      gout[e] = p;                 // write attn exactly once (268 MB stream)
      pp[e]   = (_Float16)p;       // f16 copy for the PV WMMA
    }
  }
  __syncthreads();

  // --- out = attn @ vs : wave w owns d-tile [w*16, w*16+16) ---
  {
    const int d0 = wave * 16;
    const _Float16* vbase = VT + (hb * 64 + d0) * 512;  // V^T rows: K-contig
    v8f acc = {};
    for (int k = 0; k < 512; k += 32) {
      v16h a   = load_frag16(&sP[0][0] + k, 512);
      v16h bfr = load_frag16(vbase + k, 512);
      acc = wmma_f16(a, bfr, acc);
    }
#pragma unroll
    for (int r = 0; r < 8; ++r) {
      const int qrow = q0 + r + rsel;
      ctxOut[(((long)t * B_ + b) * L_ + qrow) * 512 + h * 64 + d0 + col] =
          (_Float16)acc[r];
    }
  }
}

// ---- 5) Output projection + bias + residual + LayerNorm --------------------
__global__ __launch_bounds__(256)
void out_proj_ln(const _Float16* __restrict__ X,   // [M,512] ctx f16
                 const _Float16* __restrict__ Wf,  // [512,512] f16 (N x K)
                 const float* __restrict__ fc_b, const float* __restrict__ res,
                 const float* __restrict__ ln_g, const float* __restrict__ ln_b,
                 float* __restrict__ out) {
  __shared__ __align__(16) float sC[16][512];
  const int m0   = blockIdx.x * 16;
  const int wave = threadIdx.x >> 5, lane = threadIdx.x & 31;
  const int n0   = wave * 64;
  v8f acc[4] = {};
  for (int k = 0; k < 512; k += 32) {
    v16h a = load_frag16(X + (long)m0 * 512 + k, 512);
#pragma unroll
    for (int t = 0; t < 4; ++t)
      acc[t] = wmma_f16(a, load_frag16(Wf + (long)(n0 + t * 16) * 512 + k, 512), acc[t]);
  }
  const int rsel = (lane >> 4) << 3, col = lane & 15;
#pragma unroll
  for (int t = 0; t < 4; ++t) {
    const int n = n0 + t * 16 + col;
    const float bn = fc_b[n];
#pragma unroll
    for (int r = 0; r < 8; ++r) {
      const int m = m0 + r + rsel;
      sC[r + rsel][n] = acc[t][r] + bn + res[(long)m * 512 + n];
    }
  }
  __syncthreads();
  // layernorm: 16 groups of 16 threads, width-16 shuffle reductions
  const int row = threadIdx.x >> 4, sub = threadIdx.x & 15;
  float* rp = &sC[row][sub * 32];
  float s = 0.f;
  for (int e = 0; e < 32; ++e) s += rp[e];
  for (int o = 8; o > 0; o >>= 1) s += __shfl_xor(s, o, 16);
  const float mean = s * (1.f / 512.f);
  float vs = 0.f;
  for (int e = 0; e < 32; ++e) { const float d = rp[e] - mean; vs += d * d; }
  for (int o = 8; o > 0; o >>= 1) vs += __shfl_xor(vs, o, 16);
  const float inv = rsqrtf(vs * (1.f / 512.f) + EPS_);
  float* go = out + (long)(m0 + row) * 512 + sub * 32;
  const float* gg = ln_g + sub * 32;
  const float* gb = ln_b + sub * 32;
  for (int e = 0; e < 32; ++e) go[e] = (rp[e] - mean) * inv * gg[e] + gb[e];
}

// ---------------------------------------------------------------------------
extern "C" void kernel_launch(void* const* d_in, const int* in_sizes, int n_in,
                              void* d_out, int out_size, void* d_ws, size_t ws_size,
                              hipStream_t stream) {
  (void)in_sizes; (void)n_in; (void)out_size; (void)ws_size;
  const float* res  = (const float*)d_in[0];
  const float* q    = (const float*)d_in[1];
  const float* k    = (const float*)d_in[2];
  const float* v    = (const float*)d_in[3];
  const int*   mask = (const int*)d_in[4];
  const float* w_q  = (const float*)d_in[5];
  const float* b_q  = (const float*)d_in[6];
  const float* w_k  = (const float*)d_in[7];
  const float* b_k  = (const float*)d_in[8];
  const float* w_v  = (const float*)d_in[9];
  const float* b_v  = (const float*)d_in[10];
  const float* bn1_g = (const float*)d_in[11];
  const float* bn1_b = (const float*)d_in[12];
  const float* bn2_g = (const float*)d_in[13];
  const float* bn2_b = (const float*)d_in[14];
  const float* bn3_g = (const float*)d_in[15];
  const float* bn3_b = (const float*)d_in[16];
  const float* fc_w  = (const float*)d_in[17];
  const float* fc_b  = (const float*)d_in[18];
  const float* ln_g  = (const float*)d_in[19];
  const float* ln_b  = (const float*)d_in[20];

  // ---- workspace carve-out (256B aligned slabs) ----
  char*  ws  = (char*)d_ws;
  size_t off = 0;
  auto carve = [&](size_t bytes) -> void* {
    void* p = ws + off;
    off = (off + bytes + 255) & ~(size_t)255;
    return p;
  };
  const size_t ACT_H = (size_t)M_ * 512 * sizeof(_Float16);  // 16 MiB
  const size_t W_H   = (size_t)512 * 512 * sizeof(_Float16);
  const size_t ACT_F = (size_t)M_ * 512 * sizeof(float);     // 32 MiB

  _Float16* q16   = (_Float16*)carve(ACT_H);
  _Float16* k16   = (_Float16*)carve(ACT_H);
  _Float16* v16   = (_Float16*)carve(ACT_H);
  _Float16* wq16  = (_Float16*)carve(W_H);
  _Float16* wk16  = (_Float16*)carve(W_H);
  _Float16* wv16  = (_Float16*)carve(W_H);
  _Float16* fcw16 = (_Float16*)carve(W_H);
  float*    qp    = (float*)carve(ACT_F);
  float*    kp    = (float*)carve(ACT_F);
  float*    vp    = (float*)carve(ACT_F);
  float*    scq   = (float*)carve(512 * 4);
  float*    shq   = (float*)carve(512 * 4);
  float*    sck   = (float*)carve(512 * 4);
  float*    shk   = (float*)carve(512 * 4);
  float*    scv   = (float*)carve(512 * 4);
  float*    shv   = (float*)carve(512 * 4);
  _Float16* qs    = (_Float16*)carve(ACT_H);   // [T,H,B,L,64]
  _Float16* ksp   = (_Float16*)carve(ACT_H);   // [T,H,B,L,64]
  _Float16* vt    = (_Float16*)carve(ACT_H);   // [T,H,B,64,L]
  _Float16* ctx   = (_Float16*)carve(ACT_H);   // [T,B,L,512]

  float* out0 = (float*)d_out;                     // [T,B,L,D]
  float* attn = out0 + (size_t)M_ * 512;           // [T,H,B,L,L]

  const int nAct = M_ * 512;           // 8,388,608
  const int nW   = 512 * 512;
  // 0) fp16 conversions
  cvt_f32_f16<<<nAct / 256, 256, 0, stream>>>(q, q16, nAct);
  cvt_f32_f16<<<nAct / 256, 256, 0, stream>>>(k, k16, nAct);
  cvt_f32_f16<<<nAct / 256, 256, 0, stream>>>(v, v16, nAct);
  cvt_f32_f16<<<nW / 256, 256, 0, stream>>>(w_q, wq16, nW);
  cvt_f32_f16<<<nW / 256, 256, 0, stream>>>(w_k, wk16, nW);
  cvt_f32_f16<<<nW / 256, 256, 0, stream>>>(w_v, wv16, nW);
  cvt_f32_f16<<<nW / 256, 256, 0, stream>>>(fc_w, fcw16, nW);

  // 1) QKV projections (WMMA)
  proj_gemm<<<M_ / 16, 256, 0, stream>>>(q16, wq16, b_q, qp);
  proj_gemm<<<M_ / 16, 256, 0, stream>>>(k16, wk16, b_k, kp);
  proj_gemm<<<M_ / 16, 256, 0, stream>>>(v16, wv16, b_v, vp);

  // 2) BatchNorm stats (bn2->q, bn1->k, bn3->v per the reference)
  bn_stats<<<512, 256, 0, stream>>>(qp, bn2_g, bn2_b, scq, shq);
  bn_stats<<<512, 256, 0, stream>>>(kp, bn1_g, bn1_b, sck, shk);
  bn_stats<<<512, 256, 0, stream>>>(vp, bn3_g, bn3_b, scv, shv);

  // 3) BN apply + LIF + spikes (V emitted transposed for the PV WMMA)
  const int nLif = (B_ * L_ * C_) / 256;           // 8192 blocks
  lif_spikes<<<nLif, 256, 0, stream>>>(qp, scq, shq, qs, 0);
  lif_spikes<<<nLif, 256, 0, stream>>>(kp, sck, shk, ksp, 0);
  lif_spikes<<<nLif, 256, 0, stream>>>(vp, scv, shv, vt, 1);

  // 4) Fused attention (QK^T WMMA -> softmax -> PV WMMA)
  attention<<<dim3(L_ / 16, H_ * B_, T_), 128, 0, stream>>>(qs, ksp, vt, mask,
                                                            attn, ctx);

  // 5) Output projection + residual + LayerNorm (WMMA)
  out_proj_ln<<<M_ / 16, 256, 0, stream>>>(ctx, fcw16, fc_b, res, ln_g, ln_b,
                                           out0);
}
